// MultiHeadSelfAttention_35725537968584
// MI455X (gfx1250) — matrix operand order repaired and checked by reading, not compiled, not run on hardware
//
#include <hip/hip_runtime.h>

// ---------------------------------------------------------------------------
// Types / WMMA helpers (gfx1250, wave32)
// ---------------------------------------------------------------------------
typedef __bf16 bf16_t;
typedef __attribute__((ext_vector_type(16))) __bf16 v16bf;
typedef __attribute__((ext_vector_type(8)))  float  v8f;

#define D_MODEL 1024
#define N_HEADS 16
#define DKH     64
#define SEQ_T   2048
#define BATCH   2

union FragU { v16bf v; uint4 q[2]; };

// A-operand (16x32, 16-bit): lane L holds row M=L%16, K chunks
//   [c0, c0+8) and [c0+16, c0+24), c0 = (L<16) ? 0 : 8   (ISA 7.12.2)
static __device__ __forceinline__ v16bf load_a_frag(const bf16_t* row, int lane) {
    const int c0 = (lane < 16) ? 0 : 8;
    FragU f;
    f.q[0] = *(const uint4*)(row + c0);
    f.q[1] = *(const uint4*)(row + c0 + 16);
    return f.v;
}
// B-operand (32x16, 16-bit): lane L holds col N=L%16, K chunks
//   [k0, k0+8) and [k0+16, k0+24), k0 = (L<16) ? 0 : 8
static __device__ __forceinline__ v16bf load_b_frag(const bf16_t* col, int lane) {
    const int k0 = (lane < 16) ? 0 : 8;
    FragU f;
    f.q[0] = *(const uint4*)(col + k0);
    f.q[1] = *(const uint4*)(col + k0 + 16);
    return f.v;
}
static __device__ __forceinline__ v8f wmma_bf16(v16bf a, v16bf b, v8f c) {
    return __builtin_amdgcn_wmma_f32_16x16x32_bf16(false, a, false, b,
                                                   (short)0, c, false, false);
}
static __device__ __forceinline__ v8f vzero8() {
    v8f z = {0.f,0.f,0.f,0.f,0.f,0.f,0.f,0.f};
    return z;
}

// ---- CDNA5 async direct-to-LDS copy (16B per lane), tracked by ASYNCcnt ----
static __device__ __forceinline__ void async_copy_b128(unsigned lds_byte_off,
                                                       const bf16_t* gsrc) {
    asm volatile("global_load_async_to_lds_b128 %0, %1, off"
                 :: "v"(lds_byte_off), "v"(gsrc) : "memory");
}
static __device__ __forceinline__ void wait_async0() {
    asm volatile("s_wait_asynccnt 0" ::: "memory");
}
static __device__ __forceinline__ unsigned lds_off(const void* p) {
    return (unsigned)(size_t)p;   // flat->LDS: hardware truncates to addr[31:0]
}

// ---------------------------------------------------------------------------
// fp32 -> bf16 conversion
// ---------------------------------------------------------------------------
__global__ __launch_bounds__(256)
void cvt_f32_to_bf16(const float* __restrict__ in, bf16_t* __restrict__ out, size_t n) {
    size_t i = ((size_t)blockIdx.x * blockDim.x + threadIdx.x) * 4;
    if (i + 3 < n) {
        float4 f = *(const float4*)(in + i);
        out[i + 0] = (bf16_t)f.x;
        out[i + 1] = (bf16_t)f.y;
        out[i + 2] = (bf16_t)f.z;
        out[i + 3] = (bf16_t)f.w;
    } else {
        for (; i < n; ++i) out[i] = (bf16_t)in[i];
    }
}

// ---------------------------------------------------------------------------
// Tiled transpose: V (B,T,H*DK) bf16 -> Vt (B,H,DK,T) bf16, coalesced both ways
// grid: (T/64, B*H), block 256
// ---------------------------------------------------------------------------
__global__ __launch_bounds__(256)
void transpose_v_tiled(const bf16_t* __restrict__ V, bf16_t* __restrict__ Vt) {
    __shared__ __align__(16) bf16_t tile[64][DKH + 8];
    const int tid = threadIdx.x;
    const int bh  = blockIdx.y;
    const int b   = bh >> 4;
    const int h   = bh & (N_HEADS - 1);
    const int t0  = blockIdx.x * 64;

    // load 64 (t) x 64 (d): 2 passes of 32 rows, 8 bf16 per thread
    const int rr = tid >> 3;
    const int cc = (tid & 7) * 8;
#pragma unroll
    for (int p = 0; p < 2; ++p) {
        const int t = t0 + rr + p * 32;
        *(uint4*)&tile[rr + p * 32][cc] =
            *(const uint4*)(V + (size_t)(b * SEQ_T + t) * D_MODEL + h * DKH + cc);
    }
    __syncthreads();

    // store: row d, 16 consecutive t per thread (32B)
    const int d  = tid >> 2;
    const int tc = (tid & 3) * 16;
    union { bf16_t v[16]; uint4 q[2]; } tmp;
#pragma unroll
    for (int u = 0; u < 16; ++u) tmp.v[u] = tile[tc + u][d];
    bf16_t* dst = Vt + ((size_t)bh * DKH + d) * SEQ_T + t0 + tc;
    *(uint4*)(dst)     = tmp.q[0];
    *(uint4*)(dst + 8) = tmp.q[1];
}

// ---------------------------------------------------------------------------
// GEMM: Out[M,N] = A[M,K] * W[N,K]^T   (bf16 in, f32 acc via WMMA)
// Async double-buffered LDS pipeline; optional fused RoPE; optional bf16 out.
// Block: 256 thr = 8 waves (2x4); block tile 128x128, wave tile 64x32.
// ---------------------------------------------------------------------------
#define GT_M 128
#define GT_N 128
#define GT_K 32

__global__ __launch_bounds__(256)
void gemm_xwT_wmma(const bf16_t* __restrict__ A, const bf16_t* __restrict__ W,
                   void* __restrict__ Out, int M, int N, int K,
                   int rope, int out_bf16) {
    __shared__ __align__(16) bf16_t As[2][GT_M][GT_K + 8];
    __shared__ __align__(16) bf16_t Bs[2][GT_N][GT_K + 8];

    const int tid  = threadIdx.x;
    const int lane = tid & 31;
    const int wid  = tid >> 5;       // 0..7
    const int wrow = wid >> 2;       // 0..1 (64 rows)
    const int wcol = wid & 3;        // 0..3 (32 cols)
    const int m0 = blockIdx.x * GT_M;
    const int n0 = blockIdx.y * GT_N;

    v8f acc[4][2];
#pragma unroll
    for (int i = 0; i < 4; ++i)
#pragma unroll
        for (int j = 0; j < 2; ++j) acc[i][j] = vzero8();

    // cooperative fill: 256 thr x 32B = 128x32 bf16 per matrix tile
    const int lr = tid >> 1;
    const int lc = (tid & 1) * 16;
    const bf16_t* gA = A + (size_t)(m0 + lr) * K + lc;
    const bf16_t* gB = W + (size_t)(n0 + lr) * K + lc;

    // prologue: async-stream tile 0 into buffer 0
    {
        const unsigned la = lds_off(&As[0][lr][lc]);
        const unsigned lb = lds_off(&Bs[0][lr][lc]);
        async_copy_b128(la,      gA);
        async_copy_b128(la + 16, gA + 8);
        async_copy_b128(lb,      gB);
        async_copy_b128(lb + 16, gB + 8);
    }

    int cur = 0;
    for (int k0 = 0; k0 < K; k0 += GT_K) {
        wait_async0();          // my slices of tile k0 have landed in LDS
        __syncthreads();        // everyone's slices have landed; prev reads done

        if (k0 + GT_K < K) {    // stream next tile into the other buffer
            const int nb = cur ^ 1;
            const unsigned la = lds_off(&As[nb][lr][lc]);
            const unsigned lb = lds_off(&Bs[nb][lr][lc]);
            async_copy_b128(la,      gA + k0 + GT_K);
            async_copy_b128(la + 16, gA + k0 + GT_K + 8);
            async_copy_b128(lb,      gB + k0 + GT_K);
            async_copy_b128(lb + 16, gB + k0 + GT_K + 8);
        }

        v16bf af[4], bfm[2];
#pragma unroll
        for (int i = 0; i < 4; ++i)
            af[i] = load_a_frag(&As[cur][wrow * 64 + i * 16 + (lane & 15)][0], lane);
#pragma unroll
        for (int j = 0; j < 2; ++j)
            bfm[j] = load_b_frag(&Bs[cur][wcol * 32 + j * 16 + (lane & 15)][0], lane);

#pragma unroll
        for (int i = 0; i < 4; ++i)
#pragma unroll
            for (int j = 0; j < 2; ++j)
                acc[i][j] = wmma_bf16(af[i], bfm[j], acc[i][j]);

        cur ^= 1;
    }

    // epilogue: C layout -> (row = vgpr + 8*half, col = lane%16)
    const int half8 = (lane >= 16) ? 8 : 0;
#pragma unroll
    for (int i = 0; i < 4; ++i) {
#pragma unroll
        for (int j = 0; j < 2; ++j) {
#pragma unroll
            for (int r = 0; r < 8; ++r) {
                int row = m0 + wrow * 64 + i * 16 + r + half8;
                int col = n0 + wcol * 32 + j * 16 + (lane & 15);
                float v = acc[i][j][r];
                if (rope) {
                    float other = __shfl_xor(v, 1, 32);   // even<->odd channel
                    int   hd    = col & (DKH - 1);
                    float expo  = (float)(hd & ~1) * (1.0f / (float)DKH);
                    float freq  = __expf(-9.210340371976184f * expo); // 10000^-expo
                    int   t     = row & (SEQ_T - 1);
                    float s, c;
                    __sincosf((float)t * freq, &s, &c);
                    v = (col & 1) ? (other * s + v * c) : (v * c - other * s);
                }
                size_t off = (size_t)row * N + col;
                if (out_bf16) ((bf16_t*)Out)[off] = (bf16_t)v;
                else          ((float*)Out)[off]  = v;
            }
        }
    }
}

// ---------------------------------------------------------------------------
// Flash attention: per-wave 16 query rows, causal key blocks of 32.
// Q,K: (B,T,H*DK) bf16 (RoPE'd).  Vt: (B,H,DK,T) bf16.  O: (B,T,H*DK) f32.
// ---------------------------------------------------------------------------
__global__ __launch_bounds__(256)
void flash_attn_wmma(const bf16_t* __restrict__ Q, const bf16_t* __restrict__ Km,
                     const bf16_t* __restrict__ Vt, float* __restrict__ O) {
    __shared__ __align__(16) bf16_t Plds[8][16][GT_K + 8];  // per-wave P staging

    const int tid   = threadIdx.x;
    const int lane  = tid & 31;
    const int wid   = tid >> 5;
    const int qblk  = blockIdx.x;              // T/128
    const int bh    = blockIdx.y;              // b*H + h
    const int b     = bh >> 4;
    const int h     = bh & (N_HEADS - 1);
    const int qbase = qblk * 128 + wid * 16;
    const int half8 = (lane >= 16) ? 8 : 0;

    const bf16_t* Qh = Q  + (size_t)b * SEQ_T * D_MODEL + (size_t)h * DKH;
    const bf16_t* Kh = Km + (size_t)b * SEQ_T * D_MODEL + (size_t)h * DKH;
    const bf16_t* Vh = Vt + ((size_t)(b * N_HEADS + h) * DKH) * SEQ_T;

    // Q A-fragments for d-steps 0,1 (d_k = 64 = 2 x 32)
    const bf16_t* qrow = Qh + (size_t)(qbase + (lane & 15)) * D_MODEL;
    v16bf qa0 = load_a_frag(qrow, lane);
    v16bf qa1 = load_a_frag(qrow + 32, lane);

    float mrow[8], lrow[8];
    v8f accO[4];
#pragma unroll
    for (int r = 0; r < 8; ++r) { mrow[r] = -1e30f; lrow[r] = 0.f; }
#pragma unroll
    for (int i = 0; i < 4; ++i) accO[i] = vzero8();

    const float scale = 0.125f;                // 1/sqrt(64)
    const int   nkb   = (qbase + 47) >> 5;     // causal key-block count

    for (int kb = 0; kb < nkb; ++kb) {
        const int kb0 = kb * 32;

        // scores: 16 x 32 as two 16x16 tiles
        v8f s[2];
#pragma unroll
        for (int nt = 0; nt < 2; ++nt) {
            const bf16_t* krow =
                Kh + (size_t)(kb0 + nt * 16 + (lane & 15)) * D_MODEL;
            v8f a = vzero8();
            a = wmma_bf16(qa0, load_b_frag(krow, lane), a);
            a = wmma_bf16(qa1, load_b_frag(krow + 32, lane), a);
            s[nt] = a;
        }

        // online softmax, row-wise across the 16-lane halves
#pragma unroll
        for (int r = 0; r < 8; ++r) {
            const int tq  = qbase + r + half8;
            const int tk0 = kb0 + (lane & 15);
            float v0 = s[0][r] * scale;
            float v1 = s[1][r] * scale;
            if (tk0 > tq)      v0 = -1e30f;
            if (tk0 + 16 > tq) v1 = -1e30f;
            float mx = fmaxf(v0, v1);
#pragma unroll
            for (int o = 8; o >= 1; o >>= 1) mx = fmaxf(mx, __shfl_xor(mx, o, 32));
            float mnew = fmaxf(mrow[r], mx);
            float p0 = __expf(v0 - mnew);
            float p1 = __expf(v1 - mnew);
            float ls = p0 + p1;
#pragma unroll
            for (int o = 8; o >= 1; o >>= 1) ls += __shfl_xor(ls, o, 32);
            float alpha = __expf(mrow[r] - mnew);
            lrow[r] = lrow[r] * alpha + ls;
            mrow[r] = mnew;
#pragma unroll
            for (int i = 0; i < 4; ++i) accO[i][r] *= alpha;
            // stage P (C layout -> LDS row-major 16x32)
            const int prow = r + half8;
            Plds[wid][prow][(lane & 15)]      = (bf16_t)p0;
            Plds[wid][prow][(lane & 15) + 16] = (bf16_t)p1;
        }

        // DS ops are in-order per wave; wait + compiler barrier before re-read
        asm volatile("s_wait_dscnt 0" ::: "memory");

        v16bf pa = load_a_frag(&Plds[wid][lane & 15][0], lane);
#pragma unroll
        for (int i = 0; i < 4; ++i) {
            const bf16_t* vrow = Vh + (size_t)(i * 16 + (lane & 15)) * SEQ_T + kb0;
            accO[i] = wmma_bf16(pa, load_b_frag(vrow, lane), accO[i]);
        }
    }

    // normalize + write O (B,T,H,DK) fp32
#pragma unroll
    for (int r = 0; r < 8; ++r) {
        const float inv = 1.0f / lrow[r];
        const int   tq  = qbase + r + half8;
#pragma unroll
        for (int i = 0; i < 4; ++i) {
            const int d = i * 16 + (lane & 15);
            O[((size_t)b * SEQ_T + tq) * D_MODEL + h * DKH + d] = accO[i][r] * inv;
        }
    }
}

// ---------------------------------------------------------------------------
// Host-side orchestration
// ---------------------------------------------------------------------------
extern "C" void kernel_launch(void* const* d_in, const int* in_sizes, int n_in,
                              void* d_out, int out_size, void* d_ws, size_t ws_size,
                              hipStream_t stream) {
    const float* x  = (const float*)d_in[0];
    const float* wq = (const float*)d_in[1];
    const float* wk = (const float*)d_in[2];
    const float* wv = (const float*)d_in[3];
    const float* wo = (const float*)d_in[4];
    float* out = (float*)d_out;

    const int M = BATCH * SEQ_T;       // 4096
    const int D = D_MODEL;             // 1024

    char*  ws  = (char*)d_ws;
    size_t off = 0;
    auto alloc = [&](size_t bytes) -> void* {
        void* p = ws + off;
        off += (bytes + 255) & ~(size_t)255;
        return p;
    };
    bf16_t* xb   = (bf16_t*)alloc((size_t)M * D * 2);   // reused as attn_bf later
    bf16_t* wqb  = (bf16_t*)alloc((size_t)D * D * 2);
    bf16_t* wkb  = (bf16_t*)alloc((size_t)D * D * 2);
    bf16_t* wvb  = (bf16_t*)alloc((size_t)D * D * 2);
    bf16_t* wob  = (bf16_t*)alloc((size_t)D * D * 2);
    bf16_t* Qb   = (bf16_t*)alloc((size_t)M * D * 2);
    bf16_t* Kb   = (bf16_t*)alloc((size_t)M * D * 2);
    bf16_t* Vtb  = (bf16_t*)alloc((size_t)M * D * 2);   // (B,H,DK,T)
    float*  attn = (float*)alloc((size_t)M * D * 4);    // hosts Vb bf16 first
    bf16_t* Vb   = (bf16_t*)attn;                       // alias (dead before attn write)
    bf16_t* attn_bf = xb;                               // alias (x dead after V proj)

    const size_t nX = (size_t)M * D;
    const size_t nW = (size_t)D * D;
    dim3 cvtX((unsigned)((nX / 4 + 255) / 256));
    dim3 cvtW((unsigned)((nW / 4 + 255) / 256));
    dim3 blk(256);
    dim3 gemmGrid(M / GT_M, D / GT_N);                  // 32 x 8
    dim3 attnGrid(SEQ_T / 128, BATCH * N_HEADS);        // 16 x 32
    dim3 trGrid(SEQ_T / 64, BATCH * N_HEADS);           // 32 x 32

    // 1) bf16 conversions
    cvt_f32_to_bf16<<<cvtX, blk, 0, stream>>>(x,  xb,  nX);
    cvt_f32_to_bf16<<<cvtW, blk, 0, stream>>>(wq, wqb, nW);
    cvt_f32_to_bf16<<<cvtW, blk, 0, stream>>>(wk, wkb, nW);
    cvt_f32_to_bf16<<<cvtW, blk, 0, stream>>>(wv, wvb, nW);
    cvt_f32_to_bf16<<<cvtW, blk, 0, stream>>>(wo, wob, nW);

    // 2) projections (RoPE fused for Q,K; bf16 outputs)
    gemm_xwT_wmma<<<gemmGrid, blk, 0, stream>>>(xb, wqb, Qb, M, D, D, 1, 1);
    gemm_xwT_wmma<<<gemmGrid, blk, 0, stream>>>(xb, wkb, Kb, M, D, D, 1, 1);
    gemm_xwT_wmma<<<gemmGrid, blk, 0, stream>>>(xb, wvb, Vb, M, D, D, 0, 1);

    // 3) V -> Vt (B,H,DK,T), coalesced tiled transpose
    transpose_v_tiled<<<trGrid, blk, 0, stream>>>(Vb, Vtb);

    // 4) flash attention -> attn (f32, (B,T,H,DK)); overwrites Vb alias (dead)
    flash_attn_wmma<<<attnGrid, blk, 0, stream>>>(Qb, Kb, Vtb, attn);

    // 5) attn -> bf16 (reuse xb region), then output projection -> d_out (f32)
    cvt_f32_to_bf16<<<cvtX, blk, 0, stream>>>(attn, attn_bf, nX);
    gemm_xwT_wmma<<<gemmGrid, blk, 0, stream>>>(attn_bf, wob, out, M, D, D, 0, 0);
}
